// NeuralHMM_8186207666601
// MI455X (gfx1250) — compile-verified, matrix operand order checked by reading, not run on hardware
//
#include <hip/hip_runtime.h>
#include <hip/hip_bf16.h>
#include <math.h>

// ---------------- problem constants ----------------
#define Bn 16
#define Tn 512
#define Dn 768
#define Sn 4
#define Hn 48
#define On 48
#define HO (Hn * On)          // 2304
#define BT (Bn * Tn)          // 8192
#define RT (BT / 16)          // 512 row tiles
#define KB (Dn / 32)          // 24 k-blocks of 32

typedef __attribute__((ext_vector_type(16))) __bf16 v16bf;
typedef __attribute__((ext_vector_type(8)))  float  v8f;

// ---------------- workspace layout (bytes, 256-aligned) ----------------
#define OFF_A     ((size_t)0)                     // bf16 packed A: BT*Dn*2      = 12,582,912
#define OFF_B     ((size_t)12582912)              // bf16 packed B: S*HO*Dn*2    = 14,155,776
#define OFF_EM    ((size_t)26738688)              // emiss softmax  S*HO*4       =     36,864
#define OFF_LT    ((size_t)26775552)              // log_trans      H*H*4        =      9,216
#define OFF_LP    ((size_t)26784768)              // log prior      H*4 (pad 256)
#define OFF_LEPS  ((size_t)26785024)              // per-source lep BT*H*S*4     =  6,291,456
#define OFF_LEP   ((size_t)33076480)              // lep            BT*H*4       =  1,572,864
#define OFF_LA    ((size_t)34649344)              // log_alpha      BT*H*4
#define OFF_LB    ((size_t)36222208)              // log_beta(rolled) BT*H*4
#define OFF_TERMS ((size_t)37795072)              // per-(b,t) terms BT*4        =     32,768

// ---------------- helpers ----------------
__device__ __forceinline__ unsigned short f2bf(float f) {
  unsigned int u = __float_as_uint(f);
  u += 0x7FFFu + ((u >> 16) & 1u);   // round-to-nearest-even
  return (unsigned short)(u >> 16);
}
__device__ __forceinline__ float hmax16(float x) {
  x = fmaxf(x, __shfl_xor(x, 1, 32));
  x = fmaxf(x, __shfl_xor(x, 2, 32));
  x = fmaxf(x, __shfl_xor(x, 4, 32));
  x = fmaxf(x, __shfl_xor(x, 8, 32));
  return x;
}
__device__ __forceinline__ float hsum16(float x) {
  x += __shfl_xor(x, 1, 32);
  x += __shfl_xor(x, 2, 32);
  x += __shfl_xor(x, 4, 32);
  x += __shfl_xor(x, 8, 32);
  return x;
}

// ---------------- pack emb -> WMMA A-operand layout (bf16) ----------------
// A tile (16x32 bf16): lane<16 -> K {0..7,16..23}; lane>=16 -> K {8..15,24..31}
__global__ void k_pack_A(const float* __restrict__ emb, unsigned short* __restrict__ Ap) {
  int idx = blockIdx.x * blockDim.x + threadIdx.x;
  if (idx >= RT * KB * 32 * 16) return;
  int e    = idx & 15;
  int lane = (idx >> 4) & 31;
  int kbrt = idx >> 9;
  int kb   = kbrt % KB;
  int rt   = kbrt / KB;
  int kbase = (lane >= 16) ? 8 : 0;
  int k = kb * 32 + kbase + e + ((e >= 8) ? 8 : 0);
  int g = rt * 16 + (lane & 15);          // flattened b*T+t row
  Ap[idx] = f2bf(emb[(size_t)g * Dn + k]);
}

// ---------------- pack W_emiss -> WMMA B-operand layout (bf16) ----------------
// B tile (32x16): lanes hold N, same K striping as A. Tiles: (s,h,j) j=0..2 over O.
__global__ void k_pack_B(const float* __restrict__ W, unsigned short* __restrict__ Bp) {
  int idx = blockIdx.x * blockDim.x + threadIdx.x;
  if (idx >= Sn * Hn * 3 * KB * 32 * 16) return;
  int e    = idx & 15;
  int lane = (idx >> 4) & 31;
  int kbu  = idx >> 9;
  int kb   = kbu % KB;
  int u    = kbu / KB;
  int j    = u % 3;
  int h    = (u / 3) % Hn;
  int s    = u / (3 * Hn);
  int kbase = (lane >= 16) ? 8 : 0;
  int k = kb * 32 + kbase + e + ((e >= 8) ? 8 : 0);
  int n = h * On + j * 16 + (lane & 15);  // row of W_emiss[s, H*O, D]
  Bp[idx] = f2bf(W[((size_t)s * HO + n) * Dn + k]);
}

// ---------------- small prep: lp, log_trans, emiss softmax ----------------
__global__ void k_smallprep(const float* __restrict__ sp, const float* __restrict__ ut,
                            const float* __restrict__ ue, float* __restrict__ lp,
                            float* __restrict__ lt, float* __restrict__ em) {
  int blk = blockIdx.x, tid = threadIdx.x;
  if (tid >= 48) return;
  if (blk == 0) {                                        // log_softmax(state_priors)
    float mx = -3.4e38f; for (int i = 0; i < 48; ++i) mx = fmaxf(mx, sp[i]);
    float s = 0.f;       for (int i = 0; i < 48; ++i) s += __expf(sp[i] - mx);
    lp[tid] = sp[tid] - (mx + __logf(s));
  } else if (blk <= Hn) {                                // log_softmax rows of trans
    const float* row = ut + (blk - 1) * Hn;
    float mx = -3.4e38f; for (int i = 0; i < 48; ++i) mx = fmaxf(mx, row[i]);
    float s = 0.f;       for (int i = 0; i < 48; ++i) s += __expf(row[i] - mx);
    lt[(blk - 1) * Hn + tid] = row[tid] - (mx + __logf(s));
  } else {                                               // softmax rows of unnorm_emiss
    int r = blk - 1 - Hn;                                // (s*H+h)
    const float* row = ue + (size_t)r * On;
    float mx = -3.4e38f; for (int i = 0; i < 48; ++i) mx = fmaxf(mx, row[i]);
    float s = 0.f;       for (int i = 0; i < 48; ++i) s += __expf(row[i] - mx);
    em[(size_t)r * On + tid] = __expf(row[tid] - mx) / s;
  }
}

// ---------------- fused GEMM + softmax + logsumexp emission kernel ----------------
// one wave per (row-tile rt, source s, state h); 3 accumulators cover the 48 O cols
__global__ __launch_bounds__(128) void k_emiss(
    const unsigned short* __restrict__ Ap, const unsigned short* __restrict__ Bp,
    const float* __restrict__ bias, const float* __restrict__ em,
    const float* __restrict__ obs, float* __restrict__ lepS) {
  int wave = blockIdx.x * (blockDim.x >> 5) + (threadIdx.x >> 5);
  int lane = threadIdx.x & 31;
  int h  = wave % Hn;
  int s  = (wave / Hn) & 3;
  int rt = wave / (Hn * Sn);

  const v16bf* A  = ((const v16bf*)Ap) + (size_t)rt * (KB * 32) + lane;
  const v16bf* Bb = ((const v16bf*)Bp) + (size_t)(s * Hn + h) * (3 * KB * 32) + lane;

  v8f c0 = {}, c1 = {}, c2 = {};
  for (int kb = 0; kb < KB; ++kb) {
    v16bf a  = A[kb * 32];
    v16bf b0 = Bb[kb * 32];
    v16bf b1 = Bb[KB * 32 + kb * 32];
    v16bf b2 = Bb[2 * KB * 32 + kb * 32];
    c0 = __builtin_amdgcn_wmma_f32_16x16x32_bf16(false, a, false, b0, (short)0, c0, false, false);
    c1 = __builtin_amdgcn_wmma_f32_16x16x32_bf16(false, a, false, b1, (short)0, c1, false, false);
    c2 = __builtin_amdgcn_wmma_f32_16x16x32_bf16(false, a, false, b2, (short)0, c2, false, false);
  }

  int col = lane & 15;
  float bi0 = bias[(size_t)s * HO + h * On + col];
  float bi1 = bias[(size_t)s * HO + h * On + 16 + col];
  float bi2 = bias[(size_t)s * HO + h * On + 32 + col];
  float em0 = em[(size_t)(s * Hn + h) * On + col];
  float em1 = em[(size_t)(s * Hn + h) * On + 16 + col];
  float em2 = em[(size_t)(s * Hn + h) * On + 32 + col];
  int mhalf = (lane >> 4) << 3;                    // 0 or 8

  for (int v = 0; v < 8; ++v) {
    int m  = v + mhalf;
    int g  = rt * 16 + m;                          // flattened b*T+t
    int bb = g >> 9, t = g & (Tn - 1);
    const float* ob = obs + (((size_t)(bb * Tn + t)) * Sn + s) * On;

    float x0 = c0[v] + bi0, x1 = c1[v] + bi1, x2 = c2[v] + bi2;
    float mx = hmax16(fmaxf(x0, fmaxf(x1, x2)));
    float e0 = __expf(x0 - mx), e1 = __expf(x1 - mx), e2 = __expf(x2 - mx);
    float inv = 1.0f / hsum16(e0 + e1 + e2);
    // log( (1-w)*emiss + w*nn ) + log(obs),  w = 0.5
    float v0 = __logf(0.5f * em0 + 0.5f * e0 * inv) + __logf(ob[col]);
    float v1 = __logf(0.5f * em1 + 0.5f * e1 * inv) + __logf(ob[16 + col]);
    float v2 = __logf(0.5f * em2 + 0.5f * e2 * inv) + __logf(ob[32 + col]);
    float mm = hmax16(fmaxf(v0, fmaxf(v1, v2)));
    float ss = hsum16(__expf(v0 - mm) + __expf(v1 - mm) + __expf(v2 - mm));
    if ((lane & 15) == 0)
      lepS[(((size_t)(bb * Tn + t)) * Hn + h) * Sn + s] = mm + __logf(ss);
  }
}

// ---------------- sum lep over sources ----------------
__global__ void k_lepsum(const float* __restrict__ lepS, float* __restrict__ lep) {
  int i = blockIdx.x * blockDim.x + threadIdx.x;
  if (i >= BT * Hn) return;
  const float* p = lepS + (size_t)i * Sn;
  lep[i] = p[0] + p[1] + p[2] + p[3];
}

// ---------------- forward/backward scans (one workgroup per batch) ----------------
__global__ __launch_bounds__(64) void k_scan(const float* __restrict__ lep,
    const float* __restrict__ ltg, const float* __restrict__ lp,
    const int* __restrict__ seqlen, float* __restrict__ la, float* __restrict__ lbr) {
  __shared__ float lt[HO];
  __shared__ float prev[Hn];
  __shared__ float cur[Hn];
  int b = blockIdx.x, tid = threadIdx.x;
  for (int i = tid; i < HO; i += 64) lt[i] = ltg[i];
  __syncthreads();
  int len = seqlen[b];
  const float* lepb = lep + (size_t)b * Tn * Hn;
  float* lab = la  + (size_t)b * Tn * Hn;
  float* lbb = lbr + (size_t)b * Tn * Hn;

  // forward: a0 = lognorm(lp + lep[0])
  if (tid < Hn) cur[tid] = lp[tid] + lepb[tid];
  __syncthreads();
  if (tid < Hn) {
    float mx = -3.4e38f; for (int i = 0; i < Hn; ++i) mx = fmaxf(mx, cur[i]);
    float s = 0.f;       for (int i = 0; i < Hn; ++i) s += __expf(cur[i] - mx);
    float a = cur[tid] - (mx + __logf(s));
    prev[tid] = a; lab[tid] = a;
  }
  __syncthreads();
  for (int t = 1; t < Tn; ++t) {
    if (tid < Hn) {
      float mx = -3.4e38f;
      for (int i = 0; i < Hn; ++i) mx = fmaxf(mx, prev[i] + lt[i * Hn + tid]);
      float s = 0.f;
      for (int i = 0; i < Hn; ++i) s += __expf(prev[i] + lt[i * Hn + tid] - mx);
      cur[tid] = lepb[(size_t)t * Hn + tid] + mx + __logf(s);
    }
    __syncthreads();
    if (tid < Hn) {
      float mx = -3.4e38f; for (int i = 0; i < Hn; ++i) mx = fmaxf(mx, cur[i]);
      float s = 0.f;       for (int i = 0; i < Hn; ++i) s += __expf(cur[i] - mx);
      float a = cur[tid] - (mx + __logf(s));
      lab[(size_t)t * Hn + tid] = a; prev[tid] = a;
    }
    __syncthreads();
  }

  // backward (stored pre-rolled by (t+len)%T): beta[t][i]=lse_j(lt[i,j]+lep[t,j]+beta_next[j])
  if (tid < Hn) { prev[tid] = 0.f; lbb[(size_t)(((Tn - 1) + len) & (Tn - 1)) * Hn + tid] = 0.f; }
  __syncthreads();
  for (int t = Tn - 2; t >= 0; --t) {
    if (tid < Hn) {
      float mx = -3.4e38f;
      for (int j = 0; j < Hn; ++j)
        mx = fmaxf(mx, lt[tid * Hn + j] + lepb[(size_t)t * Hn + j] + prev[j]);
      float s = 0.f;
      for (int j = 0; j < Hn; ++j)
        s += __expf(lt[tid * Hn + j] + lepb[(size_t)t * Hn + j] + prev[j] - mx);
      cur[tid] = mx + __logf(s);
    }
    __syncthreads();
    if (tid < Hn) {
      prev[tid] = cur[tid];
      lbb[(size_t)((t + len) & (Tn - 1)) * Hn + tid] = cur[tid];
    }
    __syncthreads();
  }
}

// ---------------- per-(b,t) gamma / xi terms ----------------
__device__ __forceinline__ float blkMax(float v, float* red, int tid) {
  red[tid] = v; __syncthreads();
  for (int s = 128; s > 0; s >>= 1) { if (tid < s) red[tid] = fmaxf(red[tid], red[tid + s]); __syncthreads(); }
  float r = red[0]; __syncthreads(); return r;
}
__device__ __forceinline__ float blkSum(float v, float* red, int tid) {
  red[tid] = v; __syncthreads();
  for (int s = 128; s > 0; s >>= 1) { if (tid < s) red[tid] += red[tid + s]; __syncthreads(); }
  float r = red[0]; __syncthreads(); return r;
}

__global__ __launch_bounds__(256) void k_terms(const float* __restrict__ la,
    const float* __restrict__ lbr, const float* __restrict__ lep,
    const float* __restrict__ ltg, const float* __restrict__ lp,
    const int* __restrict__ seqlen, float* __restrict__ terms) {
  __shared__ float lt[HO];
  __shared__ float sA[Hn], sB[Hn], sL[Hn], sAp[Hn], sLp[Hn];
  __shared__ float red[256];
  int bt = blockIdx.x, tid = threadIdx.x;
  int b = bt >> 9, t = bt & (Tn - 1);
  for (int i = tid; i < HO; i += 256) lt[i] = ltg[i];
  if (tid < Hn) {
    size_t base = (size_t)bt * Hn;
    sA[tid] = la[base + tid]; sB[tid] = lbr[base + tid]; sL[tid] = lep[base + tid];
    sAp[tid] = (t > 0) ? la[base - Hn + tid] : 0.f;
    sLp[tid] = lp[tid];
  }
  __syncthreads();

  // gamma = lognorm(alpha+beta); emission + prior terms (redundant per thread)
  float mx = -3.4e38f;
  for (int h = 0; h < Hn; ++h) mx = fmaxf(mx, sA[h] + sB[h]);
  float s = 0.f;
  for (int h = 0; h < Hn; ++h) s += __expf(sA[h] + sB[h] - mx);
  float z = mx + __logf(s);
  float emis = 0.f, prior = 0.f;
  for (int h = 0; h < Hn; ++h) {
    float g = __expf(sA[h] + sB[h] - z);
    emis += g * sL[h]; prior += g * sLp[h];
  }

  int len = seqlen[b];
  float tran = 0.f;
  if (t >= 1) {   // xi_t uses alpha[t-1], lep[t]+beta[t]; norm over full HxH
    float lmx = -3.4e38f;
    for (int e = tid; e < HO; e += 256) {
      int i = e / Hn, j = e - i * Hn;
      float val = lt[e] + sAp[i] + sL[j] + sB[j];
      lmx = fmaxf(lmx, val);
    }
    float gm = blkMax(lmx, red, tid);
    float ls = 0.f, wsum = 0.f;
    for (int e = tid; e < HO; e += 256) {
      int i = e / Hn, j = e - i * Hn;
      float val = lt[e] + sAp[i] + sL[j] + sB[j];
      float ex = __expf(val - gm);
      ls += ex; wsum += ex * lt[e];
    }
    float LS = blkSum(ls, red, tid);
    float WS = blkSum(wsum, red, tid);
    tran = WS / LS;                 // sum exp(xi_norm)*log_trans
  }

  if (tid == 0) {
    float r = 0.f;
    if (t < len) { r += emis; if (t >= 1) r += tran; }
    if (t == 0) r += prior;
    terms[bt] = r;
  }
}

// ---------------- final reduction: mean over batch ----------------
__global__ __launch_bounds__(256) void k_final(const float* __restrict__ terms,
                                               float* __restrict__ out) {
  __shared__ float red[256];
  float s = 0.f;
  for (int i = threadIdx.x; i < BT; i += 256) s += terms[i];
  red[threadIdx.x] = s; __syncthreads();
  for (int st = 128; st > 0; st >>= 1) {
    if (threadIdx.x < st) red[threadIdx.x] += red[threadIdx.x + st];
    __syncthreads();
  }
  if (threadIdx.x == 0) out[0] = red[0] / (float)Bn;
}

// ---------------- launch ----------------
extern "C" void kernel_launch(void* const* d_in, const int* in_sizes, int n_in,
                              void* d_out, int out_size, void* d_ws, size_t ws_size,
                              hipStream_t stream) {
  const float* emb = (const float*)d_in[0];
  const float* obs = (const float*)d_in[1];
  const float* sp  = (const float*)d_in[2];
  const float* ut  = (const float*)d_in[3];
  const float* ue  = (const float*)d_in[4];
  const float* W   = (const float*)d_in[5];
  const float* be  = (const float*)d_in[6];
  const int*   sl  = (const int*)d_in[7];

  char* ws = (char*)d_ws;
  unsigned short* Ap  = (unsigned short*)(ws + OFF_A);
  unsigned short* Bp  = (unsigned short*)(ws + OFF_B);
  float* em    = (float*)(ws + OFF_EM);
  float* lt    = (float*)(ws + OFF_LT);
  float* lp    = (float*)(ws + OFF_LP);
  float* lepS  = (float*)(ws + OFF_LEPS);
  float* lep   = (float*)(ws + OFF_LEP);
  float* la    = (float*)(ws + OFF_LA);
  float* lbr   = (float*)(ws + OFF_LB);
  float* terms = (float*)(ws + OFF_TERMS);

  int nA = RT * KB * 32 * 16;                 // 6,291,456
  int nB = Sn * Hn * 3 * KB * 32 * 16;        // 7,077,888
  k_pack_A<<<(nA + 255) / 256, 256, 0, stream>>>(emb, Ap);
  k_pack_B<<<(nB + 255) / 256, 256, 0, stream>>>(W, Bp);
  k_smallprep<<<1 + Hn + Sn * Hn, 64, 0, stream>>>(sp, ut, ue, lp, lt, em);

  int nwaves = RT * Sn * Hn;                  // 98,304 waves, 4 waves/block
  k_emiss<<<nwaves / 4, 128, 0, stream>>>(Ap, Bp, be, em, obs, lepS);
  k_lepsum<<<(BT * Hn + 255) / 256, 256, 0, stream>>>(lepS, lep);

  k_scan<<<Bn, 64, 0, stream>>>(lep, lt, lp, sl, la, lbr);
  k_terms<<<BT, 256, 0, stream>>>(la, lbr, lep, lt, lp, sl, terms);
  k_final<<<1, 256, 0, stream>>>(terms, (float*)d_out);
}
